// NMTEncoder_67087389164220
// MI455X (gfx1250) — matrix operand order repaired and checked by reading, not compiled, not run on hardware
//
#include <hip/hip_runtime.h>
#include <hip/hip_bf16.h>
#include <cstdint>

// Problem constants (match reference)
#define BDIM  16
#define TDIM  2048
#define EDIM  256
#define HDIM  512
#define LNUM  3
#define DIDIM 1024                 // 2*H expansion
#define MROWS (BDIM * TDIM)        // 32768 rows of "tokens"

typedef __attribute__((ext_vector_type(16))) __bf16 v16bf;
typedef __attribute__((ext_vector_type(8)))  float  v8f;

union FragAB { v16bf v; unsigned u[8]; };
union Pack8  { __bf16 h[8]; uint4 q; };

__device__ __forceinline__ float softplus_f(float z) {
  // stable log(1+exp(z))
  if (z > 20.f) return z;
  return log1pf(__expf(z));          // exp underflows cleanly for very negative z
}
__device__ __forceinline__ float log_g_f(float x) {
  return (x >= 0.f) ? __logf(x + 0.5f) : -softplus_f(-x);
}

__device__ __forceinline__ void cvt8_bf16(const float4& lo, const float4& hi,
                                          Pack8& k) {
  k.h[0] = (__bf16)lo.x; k.h[1] = (__bf16)lo.y;
  k.h[2] = (__bf16)lo.z; k.h[3] = (__bf16)lo.w;
  k.h[4] = (__bf16)hi.x; k.h[5] = (__bf16)hi.y;
  k.h[6] = (__bf16)hi.z; k.h[7] = (__bf16)hi.w;
}

// ---------------------------------------------------------------------------
// Embedding gather with padding_idx=0 zero row: X0[m,:] = (tok==0) ? 0 : emb[tok]
// ---------------------------------------------------------------------------
__global__ __launch_bounds__(EDIM) void gather_embed(
    const int* __restrict__ tok, const float* __restrict__ emb,
    float* __restrict__ X0) {
  const int m = blockIdx.x;
  const int e = threadIdx.x;
  const int t = tok[m];
  X0[(size_t)m * EDIM + e] = (t == 0) ? 0.f : emb[(size_t)t * EDIM + e];
}

// ---------------------------------------------------------------------------
// Tiled bf16 WMMA GEMM:  out[m, colOff+n] = sum_k A[m,k] * W[n,k]  (+ bias[n])
// A: [M,K] fp32 row-major, W: [N,K] fp32 row-major. fp32 -> bf16 on the LDS
// staging path (float4 global loads, packed b128 LDS stores), fp32 accumulate
// in v_wmma_f32_16x16x32_bf16. Register-prefetch of the next K-tile overlaps
// global loads with WMMA compute (single LDS buffer).
// Block tile 128x128x32, 256 threads = 8 waves (wave32), wave tile 32x64.
// Requires M%128==0, N%128==0, K%32==0 (true for every shape here).
// ---------------------------------------------------------------------------
#define TS_M 128
#define TS_N 128
#define TS_K 32
#define LDS_STRIDE 40   // bf16 elems; 80B rows => 16B-aligned b128 slots

template <bool HAS_BIAS>
__global__ __launch_bounds__(256) void gemm_bf16_wmma(
    const float* __restrict__ A, const float* __restrict__ W,
    float* __restrict__ out, const float* __restrict__ bias,
    int M, int N, int K, int ldOut, int colOff) {
  __shared__ __align__(16) __bf16 As[TS_M * LDS_STRIDE];
  __shared__ __align__(16) __bf16 Bs[TS_N * LDS_STRIDE];

  const int tid   = threadIdx.x;
  const int lane  = tid & 31;
  const int wave  = tid >> 5;
  const int waveM = wave & 3;    // 4 waves tile M: 32 rows each
  const int waveN = wave >> 2;   // 2 waves tile N: 64 cols each
  const int llow  = lane & 15;
  const int lhigh = lane >> 4;   // lane half selects K sub-range / M+8

  const int m0 = blockIdx.y * TS_M;
  const int n0 = blockIdx.x * TS_N;

  // Staging: each thread owns 2 segments of 8 contiguous floats per tile.
  // Segment s: row = r0 + s*64, cols [c0, c0+8). 16B-aligned everywhere.
  const int r0 = tid >> 2;
  const int c0 = (tid & 3) * 8;

  const float* Abase = A + (size_t)m0 * K;
  const float* Wbase = W + (size_t)n0 * K;

  float4 pa[2][2], pb[2][2];

  auto prefetch = [&](int kt) {
    const float* Ag = Abase + (size_t)kt * TS_K;
    const float* Wg = Wbase + (size_t)kt * TS_K;
#pragma unroll
    for (int s = 0; s < 2; ++s) {
      const size_t ro = (size_t)(r0 + s * 64) * K + c0;
      pa[s][0] = *(const float4*)&Ag[ro];
      pa[s][1] = *(const float4*)&Ag[ro + 4];
      pb[s][0] = *(const float4*)&Wg[ro];
      pb[s][1] = *(const float4*)&Wg[ro + 4];
    }
  };

  auto stage = [&]() {
#pragma unroll
    for (int s = 0; s < 2; ++s) {
      Pack8 ka, kb;
      cvt8_bf16(pa[s][0], pa[s][1], ka);
      cvt8_bf16(pb[s][0], pb[s][1], kb);
      const int off = (r0 + s * 64) * LDS_STRIDE + c0;
      *(uint4*)&As[off] = ka.q;   // one ds_store_b128
      *(uint4*)&Bs[off] = kb.q;
    }
  };

  const v8f vzero = {0.f, 0.f, 0.f, 0.f, 0.f, 0.f, 0.f, 0.f};
  v8f acc[2][4];
#pragma unroll
  for (int mi = 0; mi < 2; ++mi)
#pragma unroll
    for (int ni = 0; ni < 4; ++ni) acc[mi][ni] = vzero;

  const int kTiles = K / TS_K;
  prefetch(0);
  for (int kt = 0; kt < kTiles; ++kt) {
    stage();
    __syncthreads();
    if (kt + 1 < kTiles) prefetch(kt + 1);  // overlap next loads with WMMAs

    // A fragment (16x32 bf16, ISA 7.12.2): lanes 0-15 row M=llow hold K 0..7
    // (VGPR0..3, pairs) and 16..23 (VGPR4..7); lanes 16-31 hold K 8..15/24..31.
    FragAB fa[2], fb[4];
#pragma unroll
    for (int mi = 0; mi < 2; ++mi) {
      const int row = waveM * 32 + mi * 16 + llow;
#pragma unroll
      for (int i = 0; i < 8; ++i) {
        const int koff = ((i < 4) ? (2 * i) : (16 + 2 * (i - 4))) + 8 * lhigh;
        fa[mi].u[i] = *(const unsigned*)&As[row * LDS_STRIDE + koff];
      }
    }
    // B fragment (32x16 bf16): same K layout as A; lanes 0-15 col N=llow hold
    // K 0..7 / 16..23, lanes 16-31 hold K 8..15 / 24..31.
#pragma unroll
    for (int ni = 0; ni < 4; ++ni) {
      const int col = waveN * 64 + ni * 16 + llow;
#pragma unroll
      for (int i = 0; i < 8; ++i) {
        const int koff = ((i < 4) ? (2 * i) : (16 + 2 * (i - 4))) + 8 * lhigh;
        fb[ni].u[i] = *(const unsigned*)&Bs[col * LDS_STRIDE + koff];
      }
    }

#pragma unroll
    for (int mi = 0; mi < 2; ++mi)
#pragma unroll
      for (int ni = 0; ni < 4; ++ni)
        acc[mi][ni] = __builtin_amdgcn_wmma_f32_16x16x32_bf16(
            false, fa[mi].v, false, fb[ni].v, (short)0, acc[mi][ni],
            false, false);
    __syncthreads();
  }

  // Epilogue. C layout: VGPR r -> (M=r, N=lane) lanes 0-15; (M=r+8, N=lane-16).
  // Row-pointer increments avoid per-store 64-bit multiplies.
  const size_t ldo = (size_t)ldOut;
#pragma unroll
  for (int mi = 0; mi < 2; ++mi) {
    int gncol[4];
    float bv[4];
#pragma unroll
    for (int ni = 0; ni < 4; ++ni) {
      gncol[ni] = waveN * 64 + ni * 16 + llow;
      bv[ni] = HAS_BIAS ? bias[n0 + gncol[ni]] : 0.f;
    }
    float* prow = out + (size_t)(m0 + waveM * 32 + mi * 16 + 8 * lhigh) * ldo
                      + colOff + n0;
#pragma unroll
    for (int r = 0; r < 8; ++r) {
#pragma unroll
      for (int ni = 0; ni < 4; ++ni)
        prow[gncol[ni]] = acc[mi][ni][r] + bv[ni];
      prow += ldo;
    }
  }
}

// ---------------------------------------------------------------------------
// Heinsen log-space minGRU scan, fused activation. G = [M, 2*DI] raw
// projection (hidden = cols 0..DI-1, gate = cols DI..2DI-1). One thread per
// (batch, channel); streaming cum-logsumexp with running (max, sum).
// Loads coalesced across d. grid = B * (DI/256), block = 256.
// ---------------------------------------------------------------------------
__global__ __launch_bounds__(256) void mingru_scan(
    const float* __restrict__ G, float* __restrict__ h) {
  const int b = blockIdx.x >> 2;                        // DI/256 == 4
  const int d = ((blockIdx.x & 3) << 8) + threadIdx.x;
  const float* gp = G + (size_t)b * TDIM * (2 * DIDIM);
  float* hp = h + (size_t)b * TDIM * DIDIM;

  float a_star = 0.f;
  float mrun = -3.4e38f;   // running max of (lv - a_star)
  float srun = 0.f;        // running sum of exp(x - mrun)
  for (int t = 0; t < TDIM; ++t) {
    const float hid  = gp[(size_t)t * (2 * DIDIM) + d];
    const float gate = gp[(size_t)t * (2 * DIDIM) + DIDIM + d];
    const float lc = -softplus_f(gate);                  // log(1-z)
    const float lv = -softplus_f(-gate) + log_g_f(hid);  // log z + log h~
    a_star += lc;
    const float x = lv - a_star;
    if (x > mrun) { srun = srun * __expf(mrun - x) + 1.f; mrun = x; }
    else          { srun += __expf(x - mrun); }
    hp[(size_t)t * DIDIM + d] = __expf(a_star + mrun + __logf(srun));
  }
}

// ---------------------------------------------------------------------------
// Reverse branch: per-sample flip of valid prefix into d_out cols 512..1023,
// zero padding beyond length. out is hidd [B,T,2H].
// ---------------------------------------------------------------------------
__global__ __launch_bounds__(256) void flip_rev(
    const float* __restrict__ y, const int* __restrict__ lens,
    float* __restrict__ out) {
  const size_t idx = (size_t)blockIdx.x * 256 + threadIdx.x;  // over M*H
  const int hh = (int)(idx & (HDIM - 1));
  const size_t bt = idx >> 9;                                 // H == 512
  const int b = (int)(bt / TDIM);
  const int t = (int)(bt % TDIM);
  const int src = lens[b] - 1 - t;
  float v = 0.f;
  if (src >= 0) v = y[((size_t)b * TDIM + src) * HDIM + hh];
  out[bt * (2 * HDIM) + HDIM + hh] = v;
}

// seq_h[b, :] = hidd.reshape(-1, 2H)[lens[b] + b*T]
__global__ __launch_bounds__(256) void seq_gather(
    const int* __restrict__ lens, const float* __restrict__ hidd,
    float* __restrict__ seq) {
  const int idx = blockIdx.x * 256 + threadIdx.x;  // B * 2H = 16384
  const int b = idx >> 10;
  const int c = idx & 1023;
  seq[idx] = hidd[((size_t)lens[b] + (size_t)b * TDIM) * (2 * HDIM) + c];
}

// ---------------------------------------------------------------------------
// Orchestration. Workspace layout (fp32), ~576 MB total:
//   bufx  [M,H]      64 MB  (input projection output; read-only per direction)
//   actb  [M,H]      64 MB  (layer ping / rev final output)
//   actc  [M,H]      64 MB  (layer pong)
//   G     [M,2*DI]  256 MB  (raw hg projection; embedding gather overlapped)
//   hbuf  [M,DI]    128 MB  (scan output)
// ---------------------------------------------------------------------------
extern "C" void kernel_launch(void* const* d_in, const int* in_sizes, int n_in,
                              void* d_out, int out_size, void* d_ws,
                              size_t ws_size, hipStream_t stream) {
  (void)in_sizes; (void)n_in; (void)out_size; (void)ws_size;
  const int*   x_src   = (const int*)d_in[0];
  const int*   x_len   = (const int*)d_in[1];
  const float* emb     = (const float*)d_in[2];
  const float* w_in    = (const float*)d_in[3];
  const float* b_in    = (const float*)d_in[4];
  const float* fwd_hg  = (const float*)d_in[5];
  const float* fwd_out = (const float*)d_in[6];
  const float* rev_hg  = (const float*)d_in[7];
  const float* rev_out = (const float*)d_in[8];
  float* out = (float*)d_out;

  float* bufx = (float*)d_ws;
  float* actb = bufx + (size_t)MROWS * HDIM;
  float* actc = actb + (size_t)MROWS * HDIM;
  float* G    = actc + (size_t)MROWS * HDIM;
  float* hbuf = G    + (size_t)MROWS * 2 * DIDIM;
  float* X0   = G;  // [M,E] overlaps G: fully consumed by GEMM0 before G is written

  // x = emb[x_source] @ w_in.T + b_in
  gather_embed<<<MROWS, EDIM, 0, stream>>>(x_src, emb, X0);
  gemm_bf16_wmma<true><<<dim3(HDIM / TS_N, MROWS / TS_M), 256, 0, stream>>>(
      X0, w_in, bufx, b_in, MROWS, HDIM, EDIM, HDIM, 0);

  for (int dir = 0; dir < 2; ++dir) {
    const float* whg = dir ? rev_hg : fwd_hg;
    const float* wot = dir ? rev_out : fwd_out;
    const float* in = bufx;
    for (int l = 0; l < LNUM; ++l) {
      // [hidden | gate] = in @ w_hg.T
      gemm_bf16_wmma<false>
          <<<dim3((2 * DIDIM) / TS_N, MROWS / TS_M), 256, 0, stream>>>(
              in, whg + (size_t)l * 2 * DIDIM * HDIM, G, nullptr,
              MROWS, 2 * DIDIM, HDIM, 2 * DIDIM, 0);
      // log-space associative scan over T
      mingru_scan<<<BDIM * (DIDIM / 256), 256, 0, stream>>>(G, hbuf);
      // out = h @ w_out.T
      const float* wl = wot + (size_t)l * HDIM * DIDIM;
      if (l < LNUM - 1) {
        float* o = (l == 0) ? actb : actc;
        gemm_bf16_wmma<false>
            <<<dim3(HDIM / TS_N, MROWS / TS_M), 256, 0, stream>>>(
                hbuf, wl, o, nullptr, MROWS, HDIM, DIDIM, HDIM, 0);
        in = o;
      } else if (dir == 0) {
        // forward final layer -> d_out cols [0,512), ld = 2H
        gemm_bf16_wmma<false>
            <<<dim3(HDIM / TS_N, MROWS / TS_M), 256, 0, stream>>>(
                hbuf, wl, out, nullptr, MROWS, HDIM, DIDIM, 2 * HDIM, 0);
      } else {
        // reverse final layer -> actb, flipped into d_out afterwards
        gemm_bf16_wmma<false>
            <<<dim3(HDIM / TS_N, MROWS / TS_M), 256, 0, stream>>>(
                hbuf, wl, actb, nullptr, MROWS, HDIM, DIDIM, HDIM, 0);
      }
    }
  }

  flip_rev<<<(MROWS * HDIM) / 256, 256, 0, stream>>>(actb, x_len, out);
  seq_gather<<<(BDIM * 2 * HDIM) / 256, 256, 0, stream>>>(
      x_len, out, out + (size_t)MROWS * 2 * HDIM);
}
